// DCNv2Block_38087769981398
// MI455X (gfx1250) — matrix operand order repaired and checked by reading, not compiled, not run on hardware
//
#include <hip/hip_runtime.h>
#include <hip/hip_bf16.h>

#define BB 2
#define CCH 256
#define HH 128
#define WW 128
#define OO 256
#define HW (HH * WW)
#define KC 2304          // C * 9
#define CHUNK_C 32
#define CHUNK_K 288      // CHUNK_C * 9
#define NCHUNK 8
#define NPIX 32          // pixels (N) per workgroup
#define SMP_STRIDE 296   // 288 + 8 halfs pad (bank-conflict avoidance)

typedef __attribute__((ext_vector_type(16))) __bf16 v16bf;
typedef __attribute__((ext_vector_type(8)))  __bf16 v8bf;
typedef __attribute__((ext_vector_type(8)))  float  v8f;

// ---------------------------------------------------------------------------
// Kernel A: f32 -> bf16 weight conversion (row-major [O][KC] = dcn_w layout)
// ---------------------------------------------------------------------------
__global__ void wcvt_kernel(const float* __restrict__ w,
                            __bf16* __restrict__ wb, int n) {
    int i = blockIdx.x * blockDim.x + threadIdx.x;
    if (i < n) wb[i] = (__bf16)w[i];
}

// ---------------------------------------------------------------------------
// Kernel B: 3x3 conv (27 out channels) producing the raw offset/mask field.
// One 16x16 pixel tile per block; x patch (18x18) and weights staged in LDS
// per 16-channel chunk.
// ---------------------------------------------------------------------------
__global__ __launch_bounds__(256) void offset_conv_kernel(
    const float* __restrict__ x, const float* __restrict__ offw,
    const float* __restrict__ offb, float* __restrict__ field) {
    __shared__ float s_x[16 * 368];   // 16 ch x (18 rows * stride 20, padded)
    __shared__ float s_wt[27 * 144];  // 27 out x 16 ch x 9 taps

    int blk  = blockIdx.x;
    int b    = blk >> 6;
    int tile = blk & 63;
    int ty0  = (tile >> 3) << 4;
    int tx0  = (tile & 7) << 4;
    int t    = threadIdx.x;
    const float* xb = x + (size_t)b * CCH * HW;

    float acc[27];
#pragma unroll
    for (int j = 0; j < 27; ++j) acc[j] = 0.f;
    int py = t >> 4, px = t & 15;

    for (int cg = 0; cg < 16; ++cg) {
        __syncthreads();
        for (int i = t; i < 16 * 324; i += 256) {
            int c = i / 324, r = i - c * 324;
            int yy = r / 18, xx = r - yy * 18;
            int gy = ty0 - 1 + yy, gx = tx0 - 1 + xx;
            float v = 0.f;
            if (gy >= 0 && gy < HH && gx >= 0 && gx < WW)
                v = xb[(size_t)(cg * 16 + c) * HW + gy * WW + gx];
            s_x[c * 368 + yy * 20 + xx] = v;
        }
        for (int i = t; i < 27 * 144; i += 256) {
            int j = i / 144, r = i - j * 144;  // r = c*9 + kk
            s_wt[i] = offw[(size_t)j * KC + cg * 144 + r];
        }
        __syncthreads();
#pragma unroll 1
        for (int c = 0; c < 16; ++c) {
            float v[9];
#pragma unroll
            for (int kk = 0; kk < 9; ++kk)
                v[kk] = s_x[c * 368 + (py + kk / 3) * 20 + (px + kk % 3)];
#pragma unroll
            for (int j = 0; j < 27; ++j) {
                float a = acc[j];
#pragma unroll
                for (int kk = 0; kk < 9; ++kk)
                    a += s_wt[j * 144 + c * 9 + kk] * v[kk];
                acc[j] = a;
            }
        }
    }
    int gy = ty0 + py, gx = tx0 + px;
    for (int j = 0; j < 27; ++j)
        field[((size_t)(b * 27 + j) * HH + gy) * WW + gx] = acc[j] + offb[j];
}

// ---------------------------------------------------------------------------
// Kernel C: deformable conv main GEMM via v_wmma_f32_16x16x32_bf16.
// Block = 32-pixel row segment x all 256 output channels.
// Per wave: 2 M-tiles x 2 N-tiles (4 accumulators) -> 4 WMMAs per K-step
// amortize each A-fragment load over two N-tiles.
// K loop: 8 chunks of 32 channels x 9 taps (K-slab 288).
// ---------------------------------------------------------------------------
__global__ __launch_bounds__(256) void dcn_wmma_kernel(
    const float* __restrict__ x, const float* __restrict__ field,
    const __bf16* __restrict__ wb, float* __restrict__ out) {
    __shared__ float  s_w4[288 * 4];            // 4 bilinear weights (mask folded)
    __shared__ int    s_a4[288 * 4];            // 4 clamped gather addrs (in-plane)
    __shared__ __bf16 s_smp[NPIX * SMP_STRIDE]; // sampled[pixel][K-slab]

    int blk = blockIdx.x;                 // grid = BB * HH * 4
    int b   = blk / (HH * 4);
    int rem = blk - b * (HH * 4);
    int y   = rem >> 2;
    int tx0 = (rem & 3) << 5;
    int t   = threadIdx.x;
    const float* xb = x + (size_t)b * CCH * HW;

    // ---- per-(pixel,tap) bilinear setup: addresses + mask-folded weights ----
    for (int i = t; i < 9 * NPIX; i += 256) {
        int kk = i >> 5, pixel = i & 31;
        int xq = tx0 + pixel;
        size_t base = (size_t)(b * 27) * HW + y * WW + xq;
        float dy = field[base + (size_t)(2 * kk) * HW];
        float dx = field[base + (size_t)(2 * kk + 1) * HW];
        float mr = field[base + (size_t)(18 + kk) * HW];
        float m  = 1.f / (1.f + __expf(-mr));  // sigmoid(mask)
        float pyf = (float)(y - 1 + kk / 3) + dy;
        float pxf = (float)(xq - 1 + kk % 3) + dx;
        float y0f = floorf(pyf), x0f = floorf(pxf);
        float fy = pyf - y0f, fx = pxf - x0f;
        int y0 = (int)y0f, x0 = (int)x0f;
        float wgt[4] = {(1.f - fy) * (1.f - fx) * m, (1.f - fy) * fx * m,
                        fy * (1.f - fx) * m, fy * fx * m};
#pragma unroll
        for (int cr = 0; cr < 4; ++cr) {
            int yy = y0 + (cr >> 1), xx = x0 + (cr & 1);
            bool ok = (yy >= 0) && (yy < HH) && (xx >= 0) && (xx < WW);
            int yc = min(max(yy, 0), HH - 1), xc = min(max(xx, 0), WW - 1);
            s_w4[i * 4 + cr] = ok ? wgt[cr] : 0.f;
            s_a4[i * 4 + cr] = yc * WW + xc;
        }
    }

    int lane = t & 31, wave = t >> 5;
    int lane_lo = lane & 15, laneHi = lane >> 4;
    int pixel = t & 31, sub = t >> 5;   // sampling decomposition
    int mbase = wave * 32;

    v8f acc00 = {}, acc01 = {}, acc10 = {}, acc11 = {};

    for (int cc = 0; cc < NCHUNK; ++cc) {
        __syncthreads();
        // ---- bilinear sample 32 channels x 9 taps x 32 pixels -> LDS bf16 ----
#pragma unroll 2
        for (int pass = 0; pass < 36; ++pass) {
            int j = pass * 8 + sub;            // 0..287 = (c_local, kk)
            int c_local = j / 9, kk = j - c_local * 9;
            int tri = (kk * NPIX + pixel) * 4;
            const float* plane = xb + (size_t)(cc * CHUNK_C + c_local) * HW;
            float v = s_w4[tri + 0] * plane[s_a4[tri + 0]] +
                      s_w4[tri + 1] * plane[s_a4[tri + 1]] +
                      s_w4[tri + 2] * plane[s_a4[tri + 2]] +
                      s_w4[tri + 3] * plane[s_a4[tri + 3]];
            s_smp[pixel * SMP_STRIDE + c_local * 9 + kk] = (__bf16)v;
        }
        __syncthreads();

        const __bf16* wslab = wb + cc * CHUNK_K;
        if (cc < NCHUNK - 1) {  // hint next weight slab (global_prefetch_b8)
            __builtin_prefetch(wslab + CHUNK_K + (size_t)(mbase + lane_lo) * KC, 0, 1);
            __builtin_prefetch(wslab + CHUNK_K + (size_t)(mbase + 16 + lane_lo) * KC, 0, 1);
        }
#pragma unroll 1
        for (int kst = 0; kst < 9; ++kst) {
            // B fragments: lanes 0-15 -> K 0..15, lanes 16-31 -> K 16..31 (contiguous k order, same as A)
            const v8bf* pb0 = (const v8bf*)&s_smp[lane_lo * SMP_STRIDE + kst * 32 + laneHi * 16];
            const v8bf* pb1 = (const v8bf*)&s_smp[(16 + lane_lo) * SMP_STRIDE + kst * 32 + laneHi * 16];
            v16bf bf0 = __builtin_shufflevector(pb0[0], pb0[1], 0, 1, 2, 3, 4, 5, 6, 7,
                                                8, 9, 10, 11, 12, 13, 14, 15);
            v16bf bf1 = __builtin_shufflevector(pb1[0], pb1[1], 0, 1, 2, 3, 4, 5, 6, 7,
                                                8, 9, 10, 11, 12, 13, 14, 15);
            // A fragments: row M = lane_lo; laneHi splits K {0..15}/{16..31}
            const v8bf* pa0 = (const v8bf*)(wslab + (size_t)(mbase + lane_lo) * KC + kst * 32);
            v16bf a0 = __builtin_shufflevector(pa0[2 * laneHi], pa0[2 * laneHi + 1],
                                               0, 1, 2, 3, 4, 5, 6, 7,
                                               8, 9, 10, 11, 12, 13, 14, 15);
            const v8bf* pa1 = (const v8bf*)(wslab + (size_t)(mbase + 16 + lane_lo) * KC + kst * 32);
            v16bf a1 = __builtin_shufflevector(pa1[2 * laneHi], pa1[2 * laneHi + 1],
                                               0, 1, 2, 3, 4, 5, 6, 7,
                                               8, 9, 10, 11, 12, 13, 14, 15);
            acc00 = __builtin_amdgcn_wmma_f32_16x16x32_bf16(
                false, a0, false, bf0, (short)0, acc00, false, false);
            acc01 = __builtin_amdgcn_wmma_f32_16x16x32_bf16(
                false, a0, false, bf1, (short)0, acc01, false, false);
            acc10 = __builtin_amdgcn_wmma_f32_16x16x32_bf16(
                false, a1, false, bf0, (short)0, acc10, false, false);
            acc11 = __builtin_amdgcn_wmma_f32_16x16x32_bf16(
                false, a1, false, bf1, (short)0, acc11, false, false);
        }
    }

    // D layout: VGPR r -> M = r (lanes 0-15) / r+8 (lanes 16-31), N = lane_lo
    float* ob = out + (size_t)(b * OO) * HW + y * WW + tx0 + lane_lo;
#pragma unroll
    for (int r = 0; r < 8; ++r) {
        int o0 = mbase + r + laneHi * 8;
        ob[(size_t)o0 * HW]             = acc00[r];
        ob[(size_t)o0 * HW + 16]        = acc01[r];
        ob[(size_t)(o0 + 16) * HW]      = acc10[r];
        ob[(size_t)(o0 + 16) * HW + 16] = acc11[r];
    }
}

// ---------------------------------------------------------------------------
extern "C" void kernel_launch(void* const* d_in, const int* in_sizes, int n_in,
                              void* d_out, int out_size, void* d_ws, size_t ws_size,
                              hipStream_t stream) {
    const float* x    = (const float*)d_in[0];
    const float* offw = (const float*)d_in[1];
    const float* offb = (const float*)d_in[2];
    const float* dcnw = (const float*)d_in[3];
    float* out = (float*)d_out;

    // workspace: [bf16 weights O*KC][f32 field B*27*H*W]
    __bf16* wb   = (__bf16*)d_ws;
    float* field = (float*)((char*)d_ws + (size_t)OO * KC * sizeof(__bf16));

    int nw = OO * KC;
    wcvt_kernel<<<(nw + 255) / 256, 256, 0, stream>>>(dcnw, wb, nw);
    offset_conv_kernel<<<BB * 64, 256, 0, stream>>>(x, offw, offb, field);
    dcn_wmma_kernel<<<BB * HH * 4, 256, 0, stream>>>(x, field, wb, out);
}
